// Der_SRec_13168369729717
// MI455X (gfx1250) — matrix operand order repaired and checked by reading, not compiled, not run on hardware
//
#include <hip/hip_runtime.h>
#include <math.h>

typedef __attribute__((ext_vector_type(16))) __bf16 v16bf;
typedef __attribute__((ext_vector_type(8)))  float  v8f;

#define Bsz    16384
#define Dd     512
#define TWOD   1024
#define M_ATT  (4 * Bsz)
#define KPAD   40            // padded bf16 K-stride in LDS: 80B rows (16B aligned)
#define BN_EPSF 1e-5f

#if defined(__gfx1250__) && __has_builtin(__builtin_amdgcn_global_load_async_to_lds_b128)
#define HAVE_ASYNC 1
#else
#define HAVE_ASYNC 0
#endif

// ---------------------------------------------------------------------------
// WMMA helpers (CDNA5: v_wmma_f32_16x16x32_bf16, wave32)
// ---------------------------------------------------------------------------
__device__ __forceinline__ v8f wmma_bf16(v16bf a, v16bf b, v8f c) {
    return __builtin_amdgcn_wmma_f32_16x16x32_bf16(false, a, false, b, (short)0, c,
                                                   false, false);
}

// A fragment 16x32 bf16 from LDS [row][KPAD]
__device__ __forceinline__ v16bf load_a_frag(const __bf16* As, int mt, int lane) {
    const int ln = lane & 15, lh = lane >> 4;
    const __bf16* r = As + (mt + ln) * KPAD + lh * 8;
    v16bf a;
#pragma unroll
    for (int i = 0; i < 8; ++i) { a[i] = r[i]; a[8 + i] = r[16 + i]; }
    return a;
}

// B fragment 32x16 bf16 from LDS [n][k]
__device__ __forceinline__ v16bf load_b_frag(const __bf16* Bs, int nt, int lane) {
    const int ln = lane & 15, lh = lane >> 4;
    const __bf16* r = Bs + (nt + ln) * KPAD + lh * 8;
    v16bf b;
#pragma unroll
    for (int i = 0; i < 8; ++i) { b[i] = r[i]; b[8 + i] = r[16 + i]; }
    return b;
}

__device__ __forceinline__ float hreduce16(float v) {
    v += __shfl_xor(v, 1, 32);
    v += __shfl_xor(v, 2, 32);
    v += __shfl_xor(v, 4, 32);
    v += __shfl_xor(v, 8, 32);
    return v;
}

// ---------------------------------------------------------------------------
// Async copy helpers (GLOBAL_LOAD_ASYNC_TO_LDS_B128, ASYNCcnt tracked)
// Builtin signature: (vector_size(16) int AS1*, vector_size(16) int AS3*, Ii, Ii)
// ---------------------------------------------------------------------------
#if HAVE_ASYNC
typedef int vs4i __attribute__((vector_size(4 * sizeof(int))));
typedef __attribute__((address_space(1))) vs4i g1v4i;
typedef __attribute__((address_space(3))) vs4i l3v4i;
__device__ __forceinline__ g1v4i* as_g(const void* p) {
    return (g1v4i*)(unsigned long long)p;
}
__device__ __forceinline__ l3v4i* as_l(void* p) {
    // generic LDS pointer: low 32 bits are the LDS offset (ISA flat aperture rule)
    return (l3v4i*)(unsigned int)(unsigned long long)p;
}
template <int N> __device__ __forceinline__ void wait_async() {
#if __has_builtin(__builtin_amdgcn_s_wait_asynccnt)
    __builtin_amdgcn_s_wait_asynccnt(N);
#else
    asm volatile("s_wait_asynccnt %0" ::"i"(N) : "memory");
#endif
}
#endif

// Stage one 64x32 bf16 tile (rows stride ldK) into LDS [row][KPAD].
// 256 x 16B segments, 2 per thread -> 2 async instructions per wave.
__device__ __forceinline__ void stage_tile(const __bf16* __restrict__ g, int ldK,
                                           int kk, __bf16* dst, int tid) {
#pragma unroll
    for (int j = 0; j < 2; ++j) {
        int seg = tid + j * 128;          // 0..255
        int row = seg >> 2, s = seg & 3;  // 4 x 16B per row
        const __bf16* gp = g + (size_t)row * ldK + kk + s * 8;
        __bf16* lp = dst + row * KPAD + s * 8;
#if HAVE_ASYNC
        __builtin_amdgcn_global_load_async_to_lds_b128(as_g(gp), as_l(lp), 0, 0);
#else
        *(uint4*)lp = *(const uint4*)gp;
#endif
    }
}

// ---------------------------------------------------------------------------
// Double-buffered WMMA core: 64x64 block tile, 4 waves, 2x2 wmma per wave.
// Ab: A rows (bf16, stride KTOT) for this block; Bb: B rows likewise.
// ---------------------------------------------------------------------------
template <int KTOT>
__device__ __forceinline__ void gemm_core(const __bf16* __restrict__ Ab,
                                          const __bf16* __restrict__ Bb,
                                          int tid, int lane, int mbase, int nbase,
                                          __bf16 As[2][64 * KPAD],
                                          __bf16 Bs[2][64 * KPAD], v8f acc[2][2]) {
    constexpr int T = KTOT / 32;
    stage_tile(Ab, KTOT, 0, As[0], tid);
    stage_tile(Bb, KTOT, 0, Bs[0], tid);
#pragma unroll 2
    for (int i = 0; i < T; ++i) {
        const int cur = i & 1;
        if (i + 1 < T) {
            stage_tile(Ab, KTOT, (i + 1) * 32, As[cur ^ 1], tid);
            stage_tile(Bb, KTOT, (i + 1) * 32, Bs[cur ^ 1], tid);
#if HAVE_ASYNC
            wait_async<4>();   // in-order retire: <=4 outstanding => tile i landed
#endif
        } else {
#if HAVE_ASYNC
            wait_async<0>();
#endif
        }
        __syncthreads();
        v16bf a0 = load_a_frag(As[cur], mbase,      lane);
        v16bf a1 = load_a_frag(As[cur], mbase + 16, lane);
        v16bf b0 = load_b_frag(Bs[cur], nbase,      lane);
        v16bf b1 = load_b_frag(Bs[cur], nbase + 16, lane);
        acc[0][0] = wmma_bf16(a0, b0, acc[0][0]);
        acc[0][1] = wmma_bf16(a0, b1, acc[0][1]);
        acc[1][0] = wmma_bf16(a1, b0, acc[1][0]);
        acc[1][1] = wmma_bf16(a1, b1, acc[1][1]);
        __syncthreads();       // all waves done reading buf before its reuse
    }
}

// ---------------------------------------------------------------------------
// init: scores <- att_b3, out <- b2
// ---------------------------------------------------------------------------
__global__ void k_init(float* __restrict__ scores, const float* __restrict__ ab3,
                       float* __restrict__ out, const float* __restrict__ b2) {
    int i = blockIdx.x * blockDim.x + threadIdx.x;
    if (i < M_ATT) scores[i] = ab3[0];
    if (i < Bsz)   out[i]    = b2[0];
}

// ---------------------------------------------------------------------------
// weights fp32 -> bf16 (one-time per launch; feeds 16B async staging)
// ---------------------------------------------------------------------------
__global__ void k_cvtw(const float* __restrict__ aw1, const float* __restrict__ aw2,
                       const float* __restrict__ w1f, __bf16* __restrict__ wb1,
                       __bf16* __restrict__ wb2, __bf16* __restrict__ wbf) {
    int i = blockIdx.x * blockDim.x + threadIdx.x;
    const int n1 = Dd * TWOD, n2 = Dd * Dd;
    if (i < n1) wb1[i] = (__bf16)aw1[i];
    if (i < n2) wb2[i] = (__bf16)aw2[i];
    if (i < n1) wbf[i] = (__bf16)w1f[i];
}

// ---------------------------------------------------------------------------
// GEMM 1: H1[4B,512] = relu( Xgather[4B,1024] @ att_w1^T + att_b1 ), bf16 out
// A staged via VGPR gather+cvt; B tile DMA'd async while A stages.
// ---------------------------------------------------------------------------
__global__ __launch_bounds__(128)
void k_gemm_att1(const int* __restrict__ nodes_u, const int* __restrict__ nodes_v,
                 const float* __restrict__ Ez_u, const float* __restrict__ Ec_u,
                 const float* __restrict__ E_u,
                 const float* __restrict__ Ez_v, const float* __restrict__ Ec_v,
                 const float* __restrict__ E_v,
                 const __bf16* __restrict__ wb1, const float* __restrict__ ab1,
                 __bf16* __restrict__ H1) {
    __shared__ __align__(16) __bf16 As[64 * KPAD];
    __shared__ __align__(16) __bf16 Bs[64 * KPAD];
    __shared__ const float* rowZC[64];
    __shared__ const float* rowFU[64];

    const int tid  = threadIdx.x;
    const int lane = tid & 31;
    const int wave = tid >> 5;
    const int r0 = blockIdx.x * 64;
    const int n0 = blockIdx.y * 64;

    if (tid < 64) {
        int r = r0 + tid;
        int g = r >> 14, b = r & (Bsz - 1);
        int n; const float *zc, *fu;
        if (g < 2) { n = nodes_u[b]; fu = E_u; zc = (g == 0) ? Ez_u : Ec_u; }
        else       { n = nodes_v[b]; fu = E_v; zc = (g == 2) ? Ez_v : Ec_v; }
        rowZC[tid] = zc + (size_t)n * Dd;
        rowFU[tid] = fu + (size_t)n * Dd;
    }
    __syncthreads();

    const int wy = wave >> 1, wx = wave & 1;
    const int mbase = wy * 32, nbase = wx * 32;
    const v8f vzero = {0.f, 0.f, 0.f, 0.f, 0.f, 0.f, 0.f, 0.f};
    v8f acc[2][2] = {{vzero, vzero}, {vzero, vzero}};

    const __bf16* Bb = wb1 + (size_t)n0 * TWOD;

    for (int kk = 0; kk < TWOD; kk += 32) {
        stage_tile(Bb, TWOD, kk, Bs, tid);  // async DMA overlaps gather below
        for (int idx = tid; idx < 64 * 32; idx += 128) {
            int row = idx >> 5, k = idx & 31;
            int kg = kk + k;
            float v = (kg < Dd) ? rowZC[row][kg] : rowFU[row][kg - Dd];
            As[row * KPAD + k] = (__bf16)v;
        }
#if HAVE_ASYNC
        wait_async<0>();
#endif
        __syncthreads();
        v16bf a0 = load_a_frag(As, mbase,      lane);
        v16bf a1 = load_a_frag(As, mbase + 16, lane);
        v16bf b0 = load_b_frag(Bs, nbase,      lane);
        v16bf b1 = load_b_frag(Bs, nbase + 16, lane);
        acc[0][0] = wmma_bf16(a0, b0, acc[0][0]);
        acc[0][1] = wmma_bf16(a0, b1, acc[0][1]);
        acc[1][0] = wmma_bf16(a1, b0, acc[1][0]);
        acc[1][1] = wmma_bf16(a1, b1, acc[1][1]);
        __syncthreads();
    }

    const int ln = lane & 15, lh = lane >> 4;
#pragma unroll
    for (int ti = 0; ti < 2; ++ti) {
#pragma unroll
        for (int tj = 0; tj < 2; ++tj) {
            int gcol = n0 + nbase + tj * 16 + ln;
            float bias = ab1[gcol];
#pragma unroll
            for (int v = 0; v < 8; ++v) {
                int grow = r0 + mbase + ti * 16 + lh * 8 + v;
                float h = acc[ti][tj][v] + bias;
                h = fmaxf(h, 0.f);
                H1[(size_t)grow * Dd + gcol] = (__bf16)h;
            }
        }
    }
}

// ---------------------------------------------------------------------------
// GEMM 2 + fused w3 dot: scores[r] += sum_c relu(H1 @ att_w2^T + ab2)[r,c]*w3[c]
// ---------------------------------------------------------------------------
__global__ __launch_bounds__(128)
void k_gemm_att2(const __bf16* __restrict__ H1, const __bf16* __restrict__ wb2,
                 const float* __restrict__ ab2, const float* __restrict__ aw3,
                 float* __restrict__ scores) {
    __shared__ __align__(16) __bf16 As[2][64 * KPAD];
    __shared__ __align__(16) __bf16 Bs[2][64 * KPAD];

    const int tid  = threadIdx.x;
    const int lane = tid & 31;
    const int wave = tid >> 5;
    const int r0 = blockIdx.x * 64;
    const int n0 = blockIdx.y * 64;
    const int wy = wave >> 1, wx = wave & 1;
    const int mbase = wy * 32, nbase = wx * 32;

    const v8f vzero = {0.f, 0.f, 0.f, 0.f, 0.f, 0.f, 0.f, 0.f};
    v8f acc[2][2] = {{vzero, vzero}, {vzero, vzero}};

    gemm_core<Dd>(H1 + (size_t)r0 * Dd, wb2 + (size_t)n0 * Dd,
                  tid, lane, mbase, nbase, As, Bs, acc);

    const int ln = lane & 15, lh = lane >> 4;
#pragma unroll
    for (int ti = 0; ti < 2; ++ti) {
#pragma unroll
        for (int v = 0; v < 8; ++v) {
            float s = 0.f;
#pragma unroll
            for (int tj = 0; tj < 2; ++tj) {
                int c = n0 + nbase + tj * 16 + ln;
                float h = acc[ti][tj][v] + ab2[c];
                h = fmaxf(h, 0.f);
                s += h * aw3[c];
            }
            s = hreduce16(s);
            if (ln == 0) {
                int grow = r0 + mbase + ti * 16 + lh * 8 + v;
                atomicAdd(&scores[grow], s);
            }
        }
    }
}

// ---------------------------------------------------------------------------
// fuse: pairwise softmax over (score_z, score_c), build UV[B,1024] bf16
// ---------------------------------------------------------------------------
__global__ void k_fuse(const float* __restrict__ scores,
                       const int* __restrict__ nodes_u, const int* __restrict__ nodes_v,
                       const float* __restrict__ Ez_u, const float* __restrict__ Ec_u,
                       const float* __restrict__ Ez_v, const float* __restrict__ Ec_v,
                       __bf16* __restrict__ UV) {
    int idx = blockIdx.x * blockDim.x + threadIdx.x;
    if (idx >= Bsz * TWOD) return;
    int b = idx >> 10, k = idx & (TWOD - 1);

    float sz, sc; const float *Z, *C; int n, kk;
    if (k < Dd) { sz = scores[b];           sc = scores[Bsz + b];
                  n = nodes_u[b]; Z = Ez_u; C = Ec_u; kk = k; }
    else        { sz = scores[2 * Bsz + b]; sc = scores[3 * Bsz + b];
                  n = nodes_v[b]; Z = Ez_v; C = Ec_v; kk = k - Dd; }
    float m  = fmaxf(sz, sc);
    float ez = __expf(sz - m), ec = __expf(sc - m);
    float wz = ez / (ez + ec);
    float val = wz * Z[(size_t)n * Dd + kk] + (1.f - wz) * C[(size_t)n * Dd + kk];
    UV[idx] = (__bf16)val;
}

// ---------------------------------------------------------------------------
// final GEMM + BN + relu + fused w2 dot
// ---------------------------------------------------------------------------
__global__ __launch_bounds__(128)
void k_gemm_final(const __bf16* __restrict__ UV, const __bf16* __restrict__ wbf,
                  const float* __restrict__ b1,
                  const float* __restrict__ gmm, const float* __restrict__ bta,
                  const float* __restrict__ mu,  const float* __restrict__ var,
                  const float* __restrict__ w2,  float* __restrict__ out) {
    __shared__ __align__(16) __bf16 As[2][64 * KPAD];
    __shared__ __align__(16) __bf16 Bs[2][64 * KPAD];

    const int tid  = threadIdx.x;
    const int lane = tid & 31;
    const int wave = tid >> 5;
    const int r0 = blockIdx.x * 64;
    const int n0 = blockIdx.y * 64;
    const int wy = wave >> 1, wx = wave & 1;
    const int mbase = wy * 32, nbase = wx * 32;

    const v8f vzero = {0.f, 0.f, 0.f, 0.f, 0.f, 0.f, 0.f, 0.f};
    v8f acc[2][2] = {{vzero, vzero}, {vzero, vzero}};

    gemm_core<TWOD>(UV + (size_t)r0 * TWOD, wbf + (size_t)n0 * TWOD,
                    tid, lane, mbase, nbase, As, Bs, acc);

    const int ln = lane & 15, lh = lane >> 4;
#pragma unroll
    for (int ti = 0; ti < 2; ++ti) {
#pragma unroll
        for (int v = 0; v < 8; ++v) {
            float s = 0.f;
#pragma unroll
            for (int tj = 0; tj < 2; ++tj) {
                int c = n0 + nbase + tj * 16 + ln;
                float h = acc[ti][tj][v] + b1[c];
                h = gmm[c] * (h - mu[c]) * __frsqrt_rn(var[c] + BN_EPSF) + bta[c];
                h = fmaxf(h, 0.f);
                s += h * w2[c];
            }
            s = hreduce16(s);
            if (ln == 0) {
                int grow = r0 + mbase + ti * 16 + lh * 8 + v;
                atomicAdd(&out[grow], s);
            }
        }
    }
}

// ---------------------------------------------------------------------------
// launcher
// ---------------------------------------------------------------------------
extern "C" void kernel_launch(void* const* d_in, const int* in_sizes, int n_in,
                              void* d_out, int out_size, void* d_ws, size_t ws_size,
                              hipStream_t stream) {
    const int*   nodes_u = (const int*)d_in[0];
    const int*   nodes_v = (const int*)d_in[1];
    const float* Ez_u = (const float*)d_in[2];
    const float* Ec_u = (const float*)d_in[3];
    const float* E_u  = (const float*)d_in[4];
    const float* Ez_v = (const float*)d_in[5];
    const float* Ec_v = (const float*)d_in[6];
    const float* E_v  = (const float*)d_in[7];
    const float* aw1  = (const float*)d_in[8];
    const float* ab1  = (const float*)d_in[9];
    const float* aw2  = (const float*)d_in[10];
    const float* ab2  = (const float*)d_in[11];
    const float* aw3  = (const float*)d_in[12];
    const float* ab3  = (const float*)d_in[13];
    const float* w1   = (const float*)d_in[14];
    const float* b1   = (const float*)d_in[15];
    const float* gmm  = (const float*)d_in[16];
    const float* bta  = (const float*)d_in[17];
    const float* mu   = (const float*)d_in[18];
    const float* var  = (const float*)d_in[19];
    const float* w2   = (const float*)d_in[20];
    const float* b2   = (const float*)d_in[21];
    float* out = (float*)d_out;

    char* ws = (char*)d_ws;
    const size_t offW1 = 0;                                    // 1 MB   bf16 att_w1
    const size_t offW2 = offW1 + (size_t)Dd * TWOD * 2;        // 0.5 MB bf16 att_w2
    const size_t offWF = offW2 + (size_t)Dd * Dd * 2;          // 1 MB   bf16 w1
    const size_t offH1 = offWF + (size_t)Dd * TWOD * 2;        // 64 MB  bf16 H1
    const size_t offSc = offH1 + (size_t)M_ATT * Dd * 2;       // 256 KB f32 scores
    const size_t offUV = offSc + (size_t)M_ATT * 4;            // 32 MB  bf16 UV
    __bf16* wb1    = (__bf16*)(ws + offW1);
    __bf16* wb2    = (__bf16*)(ws + offW2);
    __bf16* wbf    = (__bf16*)(ws + offWF);
    __bf16* H1     = (__bf16*)(ws + offH1);
    float*  scores = (float*) (ws + offSc);
    __bf16* UV     = (__bf16*)(ws + offUV);

    k_init<<<(M_ATT + 255) / 256, 256, 0, stream>>>(scores, ab3, out, b2);

    k_cvtw<<<(Dd * TWOD + 255) / 256, 256, 0, stream>>>(aw1, aw2, w1, wb1, wb2, wbf);

    k_gemm_att1<<<dim3(M_ATT / 64, Dd / 64), 128, 0, stream>>>(
        nodes_u, nodes_v, Ez_u, Ec_u, E_u, Ez_v, Ec_v, E_v, wb1, ab1, H1);

    k_gemm_att2<<<dim3(M_ATT / 64, Dd / 64), 128, 0, stream>>>(
        H1, wb2, ab2, aw3, scores);

    k_fuse<<<(Bsz * TWOD) / 256, 256, 0, stream>>>(
        scores, nodes_u, nodes_v, Ez_u, Ec_u, Ez_v, Ec_v, UV);

    k_gemm_final<<<dim3(Bsz / 64, Dd / 64), 128, 0, stream>>>(
        UV, wbf, b1, gmm, bta, mu, var, w2, out);
}
